// Model_16578573763176
// MI455X (gfx1250) — matrix operand order repaired and checked by reading, compiled, not run on hardware
//
#include <hip/hip_runtime.h>
#include <hip/hip_bf16.h>

// ---------------- model constants ----------------
#define BB        4
#define LL        8192
#define BL        (BB * LL)          // 32768 tokens
#define HH        64
#define NST       16                 // NSTATE
#define DCONV     16
#define DIN       128                // DINNER
#define HDIM      16                 // HEADDIM
#define NH        8                  // NHEADS
#define CONVD     160                // CONV_DIM
#define DPROJ     296                // D_IN_PROJ
#define CHUNK_    128
#define NCH       (LL / CHUNK_)      // 64 chunks per batch
#define DBLK      4

typedef _Float16 v16h __attribute__((ext_vector_type(16)));
typedef float    v8f  __attribute__((ext_vector_type(8)));
typedef _Float16 h2   __attribute__((ext_vector_type(2)));

// ---------------- small helpers ----------------
__device__ inline v16h zero_v16h() {
  v16h z;
#pragma unroll
  for (int i = 0; i < 16; ++i) z[i] = (_Float16)0;
  return z;
}
__device__ inline v8f zero_v8f() {
  v8f z;
#pragma unroll
  for (int i = 0; i < 8; ++i) z[i] = 0.f;
  return z;
}
__device__ inline void lds_fence_wave() {
  // wave-local LDS store->load ordering (per-wave scratch round-trip)
  asm volatile("s_wait_dscnt 0" ::: "memory");
}
__device__ inline void async_fence() {
  asm volatile("s_wait_asynccnt 0" ::: "memory");
}
// raw LDS byte offset of a generic pointer into a __shared__ array
__device__ inline unsigned lds_offset_of(const void* p) {
  return (unsigned)(unsigned long long)(uintptr_t)(
      __attribute__((address_space(3))) const void*)p;
}
// async DMA: 16 bytes global -> LDS per lane (ASYNCcnt-tracked)
__device__ inline void async_copy_b128(unsigned lds_byte_addr,
                                       const void* gaddr) {
  unsigned long long ga = (unsigned long long)(uintptr_t)gaddr;
  asm volatile("global_load_async_to_lds_b128 %0, %1, off"
               :: "v"(lds_byte_addr), "v"(ga)
               : "memory");
}
__device__ inline v8f wmma_f16(v16h a, v16h b, v8f c) {
  // D = A(16xK f16) * B(Kx16 f16) + C(16x16 f32); K=32 (pad with zeros for K=16)
  return __builtin_amdgcn_wmma_f32_16x16x32_f16(false, a, false, b, (short)0, c,
                                                false, false);
}

// A operand from row-major src[m][k] (stride elems per row). K in [0,KMAX), pad 0.
template <int KMAX>
__device__ inline v16h ldA_MK(const _Float16* src, int stride, int lane) {
  int m = lane & 15, hi = lane >> 4;
  const _Float16* row = src + m * stride;
  v16h a = zero_v16h();
#pragma unroll
  for (int v = 0; v < 4; ++v) {                 // K = hi*8 + 2v (+1)  (< 16 always)
    h2 p = *(const h2*)(row + hi * 8 + 2 * v);
    a[2 * v] = p[0];
    a[2 * v + 1] = p[1];
  }
  if (KMAX > 16) {
#pragma unroll
    for (int v = 0; v < 4; ++v) {               // K = 16 + hi*8 + 2v (+1)
      h2 p = *(const h2*)(row + 16 + hi * 8 + 2 * v);
      a[8 + 2 * v] = p[0];
      a[9 + 2 * v] = p[1];
    }
  }
  return a;
}

// A operand from src[k][m] (A = transpose of src), stride elems per k-row.
template <int KMAX>
__device__ inline v16h ldA_KM(const _Float16* src, int stride, int lane) {
  int m = lane & 15, hi = lane >> 4;
  v16h a = zero_v16h();
#pragma unroll
  for (int v = 0; v < 4; ++v) {
    int k = hi * 8 + 2 * v;
    a[2 * v] = src[k * stride + m];
    a[2 * v + 1] = src[(k + 1) * stride + m];
  }
  if (KMAX > 16) {
#pragma unroll
    for (int v = 0; v < 4; ++v) {
      int k = 16 + hi * 8 + 2 * v;
      a[8 + 2 * v] = src[k * stride + m];
      a[9 + 2 * v] = src[(k + 1) * stride + m];
    }
  }
  return a;
}

// B operand (KxN, N=16) where source is src[n][k] row-major ("B^T" layout).
template <int KMAX>
__device__ inline v16h ldB_NT(const _Float16* src, int stride, int lane) {
  int n = lane & 15, hi = lane >> 4;
  const _Float16* row = src + n * stride;
  v16h b = zero_v16h();
  if (KMAX > 0) {
#pragma unroll
    for (int v = 0; v < 8; ++v) {               // K = hi*8 + 2v (+8 if v>=4) (+1)
      int k = hi * 8 + 2 * v + (v >= 4 ? 8 : 0);
      if (k < KMAX) {
        h2 p = *(const h2*)(row + k);
        b[2 * v] = p[0];
        b[2 * v + 1] = p[1];
      }
    }
  }
  return b;
}

// B operand (KxN) where source is src[k][n] row-major, stride elems per k-row.
template <int KMAX>
__device__ inline v16h ldB_KN(const _Float16* src, int stride, int lane) {
  int n = lane & 15, hi = lane >> 4;
  v16h b = zero_v16h();
  if (KMAX > 0) {
#pragma unroll
    for (int v = 0; v < 8; ++v) {
      int k = hi * 8 + 2 * v + (v >= 4 ? 8 : 0);
      if (k < KMAX) {
        b[2 * v] = src[k * stride + n];
        b[2 * v + 1] = src[(k + 1) * stride + n];
      }
    }
  }
  return b;
}

__device__ inline float sigmoidf_(float x) { return 1.f / (1.f + expf(-x)); }
__device__ inline float softplusf_(float x) {
  return (x > 20.f) ? x : log1pf(expf(x));
}

// ================= K0: h = x * w_in (IN_CH == 1) =================
__global__ void k_win(const float* __restrict__ x, const float* __restrict__ w_in,
                      float* __restrict__ h) {
  int i = blockIdx.x * blockDim.x + threadIdx.x;
  if (i >= BL * HH) return;
  int tok = i >> 6, c = i & 63;
  h[i] = x[tok] * w_in[c];
}

// ================= K1: zx = h @ W_in_proj^T  (BLx64)*(296x64)^T ==========
__global__ void k_inproj(const float* __restrict__ hbuf,
                         const float* __restrict__ Wg,   // [296][64]
                         float* __restrict__ zx) {
  __shared__ __align__(16) _Float16 sW[304][64];   // padded N
  __shared__ __align__(16) _Float16 sH[128][64];
  int tid = threadIdx.x;
  int tok0 = blockIdx.x * 128;
  for (int i = tid; i < 304 * 64; i += 256) {
    int r = i >> 6, c = i & 63;
    sW[r][c] = (_Float16)(r < DPROJ ? Wg[r * 64 + c] : 0.f);
  }
  for (int i = tid; i < 128 * 64; i += 256) {
    int r = i >> 6, c = i & 63;
    sH[r][c] = (_Float16)hbuf[(tok0 + r) * 64 + c];
  }
  __syncthreads();

  int lane = tid & 31, wv = tid >> 5, hi = lane >> 4, n = lane & 15;
  v16h a0 = ldA_MK<32>(&sH[wv * 16][0], 64, lane);
  v16h a1 = ldA_MK<32>(&sH[wv * 16][32], 64, lane);
#pragma unroll 1
  for (int nt = 0; nt < 19; ++nt) {
    v16h b0 = ldB_NT<32>(&sW[nt * 16][0], 64, lane);
    v16h b1 = ldB_NT<32>(&sW[nt * 16][32], 64, lane);
    v8f acc = wmma_f16(a0, b0, zero_v8f());
    acc = wmma_f16(a1, b1, acc);
    int col = nt * 16 + n;
    if (col < DPROJ) {
#pragma unroll
      for (int r = 0; r < 8; ++r) {
        int m = r + 8 * hi;
        zx[(size_t)(tok0 + wv * 16 + m) * DPROJ + col] = acc[r];
      }
    }
  }
}

// ================= K2: causal depthwise conv + SiLU, dt softplus =========
__global__ void k_conv(const float* __restrict__ zx, const float* __restrict__ cw,
                       const float* __restrict__ cb, const float* __restrict__ dtb,
                       float* __restrict__ xs, float* __restrict__ Bm,
                       float* __restrict__ Cm, float* __restrict__ dt) {
  long i = (long)blockIdx.x * blockDim.x + threadIdx.x;
  if (i >= (long)BL * 168) return;
  int tok = (int)(i / 168), ch = (int)(i % 168);
  int b = tok / LL, l = tok % LL;
  if (ch < CONVD) {
    float s = cb[ch];
#pragma unroll
    for (int j = 0; j < DCONV; ++j) {
      int ll2 = l - (DCONV - 1) + j;
      if (ll2 >= 0)
        s += cw[ch * DCONV + j] * zx[(size_t)(b * LL + ll2) * DPROJ + DIN + ch];
    }
    float v = s * sigmoidf_(s);
    if (ch < DIN) xs[(size_t)tok * DIN + ch] = v;
    else if (ch < DIN + NST) Bm[(size_t)tok * NST + (ch - DIN)] = v;
    else Cm[(size_t)tok * NST + (ch - DIN - NST)] = v;
  } else {
    int hh = ch - CONVD;
    float v = zx[(size_t)tok * DPROJ + DIN + CONVD + hh] + dtb[hh];
    dt[(size_t)tok * NH + hh] = softplusf_(v);
  }
}

// ================= K3a: per-chunk inclusive cumsum of dt*A ===============
__global__ void k_acs(const float* __restrict__ dt, const float* __restrict__ alog,
                      float* __restrict__ acs, float* __restrict__ alast) {
  __shared__ float s[CHUNK_];
  int bid = blockIdx.x;
  int c = bid & (NCH - 1);
  int hh = (bid >> 6) & (NH - 1);
  int b = bid >> 9;
  float A = -expf(alog[hh]);
  int l = threadIdx.x;
  int tok = b * LL + c * CHUNK_ + l;
  s[l] = dt[(size_t)tok * NH + hh] * A;
  __syncthreads();
  for (int off = 1; off < CHUNK_; off <<= 1) {
    float v = (l >= off) ? s[l - off] : 0.f;
    __syncthreads();
    s[l] += v;
    __syncthreads();
  }
  acs[(size_t)(b * NH + hh) * LL + c * CHUNK_ + l] = s[l];
  if (l == CHUNK_ - 1) alast[(b * NH + hh) * NCH + c] = s[l];
}

// ================= K3b: chunk-local states = (X*dt*decay)^T @ B ==========
__global__ void k_states(const float* __restrict__ xs, const float* __restrict__ dt,
                         const float* __restrict__ Bm, const float* __restrict__ acs,
                         const float* __restrict__ alast,
                         float* __restrict__ states) {
  __shared__ __align__(16) _Float16 sXd[CHUNK_][DIN];  // 32 KB
  __shared__ __align__(16) _Float16 sB[CHUNK_][NST];   // 4 KB
  int c = blockIdx.x & (NCH - 1);
  int b = blockIdx.x >> 6;
  int tid = threadIdx.x;
  int tokbase = b * LL + c * CHUNK_;
  for (int i = tid; i < CHUNK_ * DIN; i += 256) {
    int l = i >> 7, ch = i & 127, head = ch >> 4;
    float al = alast[(b * NH + head) * NCH + c];
    float ac = acs[(size_t)(b * NH + head) * LL + c * CHUNK_ + l];
    float dec = expf(al - ac);  // <= 1
    sXd[l][ch] = (_Float16)(xs[(size_t)(tokbase + l) * DIN + ch] *
                            dt[(size_t)(tokbase + l) * NH + head] * dec);
  }
  for (int i = tid; i < CHUNK_ * NST; i += 256) {
    int l = i >> 4, n = i & 15;
    sB[l][n] = (_Float16)Bm[(size_t)(tokbase + l) * NST + n];
  }
  __syncthreads();

  int lane = tid & 31, hh = tid >> 5, hi = lane >> 4, n = lane & 15;
  v8f acc = zero_v8f();
#pragma unroll
  for (int kt = 0; kt < 4; ++kt) {
    v16h a = ldA_KM<32>(&sXd[kt * 32][hh * HDIM], DIN, lane);
    v16h bb = ldB_KN<32>(&sB[kt * 32][0], NST, lane);
    acc = wmma_f16(a, bb, acc);
  }
  size_t base = ((size_t)(b * NCH + c) * NH + hh) * 256;
#pragma unroll
  for (int r = 0; r < 8; ++r) {
    int p = r + 8 * hi;
    states[base + p * 16 + n] = acc[r];
  }
}

// ================= K3c: inter-chunk state recurrence (sequential) ========
__global__ void k_scan(const float* __restrict__ states,
                       const float* __restrict__ alast,
                       const float* __restrict__ init,  // [NH][16][16] (per block)
                       float* __restrict__ prevs) {
  int hh = blockIdx.x & (NH - 1);
  int b = blockIdx.x >> 3;
  int e = threadIdx.x;  // p*16+n
  float p = init[hh * 256 + e];
  for (int c = 0; c < NCH; ++c) {
    prevs[((size_t)(b * NCH + c) * NH + hh) * 256 + e] = p;
    float dec = expf(alast[(b * NH + hh) * NCH + c]);
    p = p * dec + states[((size_t)(b * NCH + c) * NH + hh) * 256 + e];
  }
}

// ================= K3d: Y = Y_diag + Y_off (heavy WMMA kernel) ===========
__global__ void k_ydiag(const float* __restrict__ xs, const float* __restrict__ dt,
                        const float* __restrict__ Bm, const float* __restrict__ Cm,
                        const float* __restrict__ acs,
                        const float* __restrict__ prevs,
                        float* __restrict__ ybuf) {
  __shared__ __align__(16) _Float16 sC[CHUNK_][NST];        // 4 KB
  __shared__ __align__(16) _Float16 sBm[CHUNK_][NST];       // 4 KB
  __shared__ __align__(16) _Float16 sXd[CHUNK_][DIN];       // 32 KB (X = xh*dt)
  __shared__ float sAcs[NH][CHUNK_];                        // 4 KB
  __shared__ __align__(16) _Float16 sPrevT[NH][NST][HDIM];  // 4 KB  [h][n][p]
  __shared__ __align__(16) _Float16 sScr[NH][16][16];       // 4 KB  per-wave scratch
  int c = blockIdx.x & (NCH - 1);
  int b = blockIdx.x >> 6;
  int tid = threadIdx.x;
  int tokbase = b * LL + c * CHUNK_;
  for (int i = tid; i < CHUNK_ * NST; i += 256) {
    int l = i >> 4, n = i & 15;
    sC[l][n] = (_Float16)Cm[(size_t)(tokbase + l) * NST + n];
    sBm[l][n] = (_Float16)Bm[(size_t)(tokbase + l) * NST + n];
  }
  for (int i = tid; i < CHUNK_ * DIN; i += 256) {
    int l = i >> 7, ch = i & 127, head = ch >> 4;
    sXd[l][ch] = (_Float16)(xs[(size_t)(tokbase + l) * DIN + ch] *
                            dt[(size_t)(tokbase + l) * NH + head]);
  }
  for (int i = tid; i < NH * CHUNK_; i += 256) {
    int head = i >> 7, l = i & 127;
    sAcs[head][l] = acs[(size_t)(b * NH + head) * LL + c * CHUNK_ + l];
  }
  for (int i = tid; i < NH * 256; i += 256) {
    int head = i >> 8, e = i & 255, p = e >> 4, n = e & 15;
    sPrevT[head][n][p] =
        (_Float16)prevs[((size_t)(b * NCH + c) * NH + head) * 256 + e];
  }
  __syncthreads();

  int lane = tid & 31, hh = tid >> 5, hi = lane >> 4, n = lane & 15;
  v8f Y[8];
#pragma unroll
  for (int mt = 0; mt < 8; ++mt) Y[mt] = zero_v8f();

  // ---- Y_off: (exp(Acs[l]) * C) @ prev^T ----
  v16h bprev = ldB_KN<16>(&sPrevT[hh][0][0], HDIM, lane);
  {
    int m = lane & 15;
#pragma unroll 1
    for (int mt = 0; mt < 8; ++mt) {
      float sc = expf(sAcs[hh][mt * 16 + m]);  // <= 1
      const _Float16* row = &sC[mt * 16 + m][0];
      v16h a = zero_v16h();
#pragma unroll
      for (int v = 0; v < 4; ++v) {
        int k = hi * 8 + 2 * v;  // < 16
        a[2 * v] = (_Float16)((float)row[k] * sc);
        a[2 * v + 1] = (_Float16)((float)row[k + 1] * sc);
      }
      Y[mt] = wmma_f16(a, bprev, Y[mt]);
    }
  }

  // ---- Y_diag: for each s-block, T = C@B^T masked by exp(segsum); Y += T@X ----
#pragma unroll 1
  for (int sb = 0; sb < 8; ++sb) {
    v16h bB = ldB_NT<16>(&sBm[sb * 16][0], NST, lane);
    v16h bX = ldB_KN<16>(&sXd[sb * 16][hh * HDIM], DIN, lane);
    int si = sb * 16 + n;
    float acs_s = sAcs[hh][si];
#pragma unroll 1
    for (int mt = sb; mt < 8; ++mt) {
      v16h aC = ldA_MK<16>(&sC[mt * 16][0], NST, lane);
      v8f T = wmma_f16(aC, bB, zero_v8f());
#pragma unroll
      for (int r = 0; r < 8; ++r) {
        int m = r + 8 * hi;
        int li = mt * 16 + m;
        float w = (si <= li) ? expf(sAcs[hh][li] - acs_s) : 0.f;
        sScr[hh][m][n] = (_Float16)(T[r] * w);
      }
      lds_fence_wave();
      v16h aT = ldA_MK<16>(&sScr[hh][0][0], 16, lane);
      Y[mt] = wmma_f16(aT, bX, Y[mt]);
    }
  }

  // ---- store ----
#pragma unroll
  for (int mt = 0; mt < 8; ++mt) {
#pragma unroll
    for (int r = 0; r < 8; ++r) {
      int m = r + 8 * hi;
      ybuf[(size_t)(tokbase + mt * 16 + m) * DIN + hh * HDIM + n] = Y[mt][r];
    }
  }
}

// ================= K4: skip + gate (SiLU(z)) + RMSNorm ===================
__global__ void k_gate(const float* __restrict__ ybuf, const float* __restrict__ xs,
                       const float* __restrict__ zx, const float* __restrict__ dskip,
                       const float* __restrict__ nw, _Float16* __restrict__ yn) {
  __shared__ float red[4];
  int tok = blockIdx.x;
  int ch = threadIdx.x;  // 128 threads
  int head = ch >> 4;
  float y = ybuf[(size_t)tok * DIN + ch] + xs[(size_t)tok * DIN + ch] * dskip[head];
  float zv = zx[(size_t)tok * DPROJ + ch];
  y *= zv * sigmoidf_(zv);
  float s = y * y;
#pragma unroll
  for (int o = 16; o > 0; o >>= 1) s += __shfl_xor(s, o);
  int lane = ch & 31, wid = ch >> 5;
  if (lane == 0) red[wid] = s;
  __syncthreads();
  float tot = red[0] + red[1] + red[2] + red[3];
  float scale = rsqrtf(tot * (1.f / DIN) + 1e-5f) * nw[ch];
  yn[(size_t)tok * DIN + ch] = (_Float16)(y * scale);
}

// ================= K5a: out_proj GEMM + t = y/sqrt(1+y^2) ================
__global__ void k_outproj(const _Float16* __restrict__ yn,
                          const float* __restrict__ Wo,  // [64][128]
                          _Float16* __restrict__ tbuf) {
  __shared__ __align__(16) _Float16 sY[128][DIN];  // 32 KB
  __shared__ __align__(16) _Float16 sW[64][DIN];   // 16 KB
  int tid = threadIdx.x;
  int tok0 = blockIdx.x * 128;
  // async DMA the f16 activation tile straight into LDS (ASYNCcnt)
  {
    unsigned ldsbase = lds_offset_of(&sY[0][0]);
    const char* gsrc = (const char*)(yn + (size_t)tok0 * DIN);
#pragma unroll
    for (int it = 0; it < 8; ++it) {
      int byte = (it * 256 + tid) * 16;  // 32 KB total
      async_copy_b128(ldsbase + byte, gsrc + byte);
    }
  }
  for (int i = tid; i < 64 * DIN; i += 256) {
    int r = i >> 7, cc = i & 127;
    sW[r][cc] = (_Float16)Wo[r * DIN + cc];
  }
  async_fence();
  __syncthreads();
  int lane = tid & 31, wv = tid >> 5, hi = lane >> 4, n = lane & 15;
  v16h a[4];
#pragma unroll
  for (int kt = 0; kt < 4; ++kt) a[kt] = ldA_MK<32>(&sY[wv * 16][kt * 32], DIN, lane);
#pragma unroll 1
  for (int nt = 0; nt < 4; ++nt) {
    v8f acc = zero_v8f();
#pragma unroll
    for (int kt = 0; kt < 4; ++kt) {
      v16h bb = ldB_NT<32>(&sW[nt * 16][kt * 32], DIN, lane);
      acc = wmma_f16(a[kt], bb, acc);
    }
#pragma unroll
    for (int r = 0; r < 8; ++r) {
      int m = r + 8 * hi;
      float v = acc[r];
      float t = v * rsqrtf(1.f + v * v);
      tbuf[(size_t)(tok0 + wv * 16 + m) * HH + nt * 16 + n] = (_Float16)t;
    }
  }
}

// ================= K5b: mlp GEMM + bias + residual into h ================
__global__ void k_mlp(const _Float16* __restrict__ tbuf,
                      const float* __restrict__ Wm,  // [64][64]
                      const float* __restrict__ mb, float* __restrict__ hbuf) {
  __shared__ __align__(16) _Float16 sT[128][HH];  // 16 KB
  __shared__ __align__(16) _Float16 sW[64][HH];   // 8 KB
  int tid = threadIdx.x;
  int tok0 = blockIdx.x * 128;
  // async DMA the f16 t-tile straight into LDS (ASYNCcnt)
  {
    unsigned ldsbase = lds_offset_of(&sT[0][0]);
    const char* gsrc = (const char*)(tbuf + (size_t)tok0 * HH);
#pragma unroll
    for (int it = 0; it < 4; ++it) {
      int byte = (it * 256 + tid) * 16;  // 16 KB total
      async_copy_b128(ldsbase + byte, gsrc + byte);
    }
  }
  for (int i = tid; i < 64 * HH; i += 256) {
    int r = i >> 6, cc = i & 63;
    sW[r][cc] = (_Float16)Wm[r * HH + cc];
  }
  async_fence();
  __syncthreads();
  int lane = tid & 31, wv = tid >> 5, hi = lane >> 4, n = lane & 15;
  v16h a0 = ldA_MK<32>(&sT[wv * 16][0], HH, lane);
  v16h a1 = ldA_MK<32>(&sT[wv * 16][32], HH, lane);
#pragma unroll 1
  for (int nt = 0; nt < 4; ++nt) {
    v16h b0 = ldB_NT<32>(&sW[nt * 16][0], HH, lane);
    v16h b1 = ldB_NT<32>(&sW[nt * 16][32], HH, lane);
    v8f acc = wmma_f16(a0, b0, zero_v8f());
    acc = wmma_f16(a1, b1, acc);
    int col = nt * 16 + n;
    float bias = mb[col];
#pragma unroll
    for (int r = 0; r < 8; ++r) {
      int m = r + 8 * hi;
      size_t idx = (size_t)(tok0 + wv * 16 + m) * HH + col;
      hbuf[idx] = hbuf[idx] + acc[r] + bias;
    }
  }
}

// ================= K6: out = h @ w_out^T (OUT_CH == 1) ===================
__global__ void k_wout(const float* __restrict__ hbuf,
                       const float* __restrict__ w_out, float* __restrict__ out) {
  int tok = blockIdx.x * blockDim.x + threadIdx.x;
  if (tok >= BL) return;
  float s = 0.f;
#pragma unroll
  for (int cc = 0; cc < HH; ++cc) s += hbuf[(size_t)tok * HH + cc] * w_out[cc];
  out[tok] = s;
}

// ================= host launch ==========================================
extern "C" void kernel_launch(void* const* d_in, const int* in_sizes, int n_in,
                              void* d_out, int out_size, void* d_ws, size_t ws_size,
                              hipStream_t stream) {
  (void)in_sizes; (void)n_in; (void)out_size; (void)ws_size;
  const float* x = (const float*)d_in[0];
  const float* w_in = (const float*)d_in[1];
  const float* w_out = (const float*)d_in[2];
  const float* in_proj_w = (const float*)d_in[3];
  const float* conv_w = (const float*)d_in[4];
  const float* conv_b = (const float*)d_in[5];
  const float* dt_bias = (const float*)d_in[6];
  const float* A_log = (const float*)d_in[7];
  const float* D_skip = (const float*)d_in[8];
  const float* init_states = (const float*)d_in[9];
  const float* norm_w = (const float*)d_in[10];
  const float* out_proj_w = (const float*)d_in[11];
  const float* mlp_w = (const float*)d_in[12];
  const float* mlp_b = (const float*)d_in[13];
  float* out = (float*)d_out;

  char* ws = (char*)d_ws;
  size_t off = 0;
  auto alloc = [&](size_t bytes) -> void* {
    off = (off + 255) & ~(size_t)255;
    void* p = ws + off;
    off += bytes;
    return p;
  };
  float* hbuf  = (float*)alloc((size_t)BL * HH * 4);
  float* zx    = (float*)alloc((size_t)BL * DPROJ * 4);
  float* xs    = (float*)alloc((size_t)BL * DIN * 4);
  float* Bmb   = (float*)alloc((size_t)BL * NST * 4);
  float* Cmb   = (float*)alloc((size_t)BL * NST * 4);
  float* dtb   = (float*)alloc((size_t)BL * NH * 4);
  float* acs   = (float*)alloc((size_t)BB * NH * LL * 4);
  float* alast = (float*)alloc((size_t)BB * NH * NCH * 4);
  float* states= (float*)alloc((size_t)BB * NCH * NH * 256 * 4);
  float* prevs = (float*)alloc((size_t)BB * NCH * NH * 256 * 4);
  float* ybuf  = (float*)alloc((size_t)BL * DIN * 4);
  _Float16* yn   = (_Float16*)alloc((size_t)BL * DIN * 2);
  _Float16* tbuf = (_Float16*)alloc((size_t)BL * HH * 2);

  k_win<<<(BL * HH) / 256, 256, 0, stream>>>(x, w_in, hbuf);

  for (int blk = 0; blk < DBLK; ++blk) {
    const float* Wp = in_proj_w + (size_t)blk * DPROJ * HH;
    k_inproj<<<BL / 128, 256, 0, stream>>>(hbuf, Wp, zx);
    k_conv<<<((long)BL * 168 + 255) / 256, 256, 0, stream>>>(
        zx, conv_w + (size_t)blk * CONVD * DCONV, conv_b + (size_t)blk * CONVD,
        dt_bias + blk * NH, xs, Bmb, Cmb, dtb);
    k_acs<<<BB * NH * NCH, CHUNK_, 0, stream>>>(dtb, A_log + blk * NH, acs, alast);
    k_states<<<BB * NCH, 256, 0, stream>>>(xs, dtb, Bmb, acs, alast, states);
    k_scan<<<BB * NH, 256, 0, stream>>>(states, alast,
                                        init_states + (size_t)blk * NH * 256, prevs);
    k_ydiag<<<BB * NCH, 256, 0, stream>>>(xs, dtb, Bmb, Cmb, acs, prevs, ybuf);
    k_gate<<<BL, DIN, 0, stream>>>(ybuf, xs, zx, D_skip + blk * NH,
                                   norm_w + (size_t)blk * DIN, yn);
    k_outproj<<<BL / 128, 256, 0, stream>>>(yn, out_proj_w + (size_t)blk * HH * DIN,
                                            tbuf);
    k_mlp<<<BL / 128, 256, 0, stream>>>(tbuf, mlp_w + (size_t)blk * HH * HH,
                                        mlp_b + blk * HH, hbuf);
  }
  k_wout<<<BL / 256, 256, 0, stream>>>(hbuf, w_out, out);
}
